// PhysicsFormer_30537217475131
// MI455X (gfx1250) — compile-verified, matrix operand order checked by reading, not run on hardware
//
#include <hip/hip_runtime.h>

// ---------------------------------------------------------------------------
// MI455X (gfx1250) implementation.
//  K1 k_seed_iterate : per-(batch,pair) workgroup. Stages the two 128x128 f32
//     input channels into LDS with gfx1250 async global->LDS copies
//     (global_load_async_to_lds_b128 + s_wait_asynccnt), computes
//     x = relu(tanh(0.5*(b-a))) via the exact half-pixel bilinear stencil
//     (weights 0.25/0.75), seeds cur = -x with 1.0 borders, then runs the 256
//     maxpool3x3*x iterations entirely in LDS using fp16 ping-pong buffers
//     (2 x 128KB, fits the 320KB WGP LDS) and packed v_pk_max_num_f16 /
//     v_pk_mul_f16 math. One barrier per iteration. HBM cost collapses from
//     ~68GB (256 iters from memory) to one 21MB input read + 128MB of fp16
//     intermediates that stay in the 192MB global L2.
//  K2 k_norm_l1 : per-wave 16 pixels. Channel sum-of-squares via the
//     v_wmma_f32_16x16x32_f16 Grammian trick (A == B => D = A*A^T, diagonal
//     = per-pixel ||cur||^2 over the 32 channels), one ds_bpermute to fetch
//     each lane's diagonal, then sum x*|1 - cur/max(norm,1e-12)| (valid since
//     x >= 0), wave-reduce, atomicAdd into a single f32 accumulator.
//  K3 k_finalize : clamp to FP16_MAX, NaN->0, cast to fp16 scalar.
// Workspace layout (needs 128MiB + 4B):
//   [0,64MiB)   x   fp16 [16][32][256][256]
//   [64,128MiB) cur fp16 [16][32][256][256]
//   [128MiB]    f32 accumulator
// ---------------------------------------------------------------------------

typedef _Float16 h2   __attribute__((ext_vector_type(2)));
typedef _Float16 v16h __attribute__((ext_vector_type(16)));
typedef float    v8f  __attribute__((ext_vector_type(8)));

#define PKMAX(a, b) __builtin_elementwise_max((a), (b))   // -> v_pk_max_num_f16

__constant__ int dP0[32] = {0,0,0,0,0,0,0,0,1,1,1,1,1,1,2,2,2,2,3,3,3,4,4,5,7,7,8,10,10,11,13,15};
__constant__ int dP1[32] = {1,2,3,4,5,6,7,8,2,3,4,5,6,7,3,4,5,6,4,5,6,5,6,6,8,9,9,11,12,12,14,16};

// jax.image.resize(bilinear, antialias=False), 128 -> 256, half-pixel centers:
// in_coord = o*0.5 - 0.25 ; even o=2m: 0.25*in[m-1] + 0.75*in[m] (clamped)
//                           odd  o=2m+1: 0.75*in[m] + 0.25*in[m+1] (clamped)
__device__ __forceinline__ void lincoef(int o, int& j0, int& j1, float& w0, float& w1) {
    int m = o >> 1;
    if (o & 1) { j0 = m; j1 = (m + 1 > 127) ? 127 : m + 1; w0 = 0.75f; w1 = 0.25f; }
    else       { j0 = (m - 1 < 0) ? 0 : m - 1; j1 = m;     w0 = 0.25f; w1 = 0.75f; }
}

__global__ __launch_bounds__(1024)
void k_seed_iterate(const float* __restrict__ in,
                    _Float16* __restrict__ ws_x, _Float16* __restrict__ ws_cur)
{
    extern __shared__ char smem[];
    h2*    buf0 = (h2*)smem;                 // 256 rows x 128 h2 = 128KB
    h2*    buf1 = buf0 + 256 * 128;          // second 128KB
    float* stg  = (float*)buf1;              // staging overlaps buf1 (128KB)

    const int bp  = blockIdx.x;              // 0..511 = b*32 + pair
    const int b   = bp >> 5;
    const int pr  = bp & 31;
    const int tid = threadIdx.x;
    const int r   = tid >> 2;                // output row 0..255
    const int q   = tid & 3;                 // quarter of the row
    const int c0  = q * 32;                  // first h2 column pair

    // ---- async-stage the two input channels (128x128 f32 each) into LDS ----
    // No static LDS in this module, so the dynamic-LDS base byte offset is 0:
    // sA lives at LDS bytes [128KB,192KB), sB at [192KB,256KB).
    const float* inA = in + ((size_t)b * 19 + dP0[pr]) * (128 * 128);
    const float* inB = in + ((size_t)b * 19 + dP1[pr]) * (128 * 128);
    float* sA = stg;
    float* sB = stg + 128 * 128;
    {
        const unsigned LDS_A = 131072u;             // byte offset of sA
        const unsigned LDS_B = 131072u + 65536u;    // byte offset of sB
        for (int i = tid; i < 4096; i += 1024) {    // 4096 x 16B per channel
            unsigned la = LDS_A + (unsigned)i * 16u;
            unsigned lb = LDS_B + (unsigned)i * 16u;
            const void* ga = (const char*)inA + (size_t)i * 16u;
            const void* gb = (const char*)inB + (size_t)i * 16u;
            asm volatile("global_load_async_to_lds_b128 %0, %1, off"
                         :: "v"(la), "v"(ga) : "memory");
            asm volatile("global_load_async_to_lds_b128 %0, %1, off"
                         :: "v"(lb), "v"(gb) : "memory");
        }
        asm volatile("s_wait_asynccnt 0" ::: "memory");
    }
    __syncthreads();

    // ---- bilinear resize + pair activation, kept in registers ----
    int i0, i1; float wy0, wy1;
    lincoef(r, i0, i1, wy0, wy1);
    h2 xh[32];
    #pragma unroll
    for (int i = 0; i < 32; ++i) {
        float vx[2];
        #pragma unroll
        for (int s = 0; s < 2; ++s) {
            int o = (c0 + i) * 2 + s;
            int j0, j1; float wx0, wx1;
            lincoef(o, j0, j1, wx0, wx1);
            float av = wy0 * (wx0 * sA[i0 * 128 + j0] + wx1 * sA[i0 * 128 + j1])
                     + wy1 * (wx0 * sA[i1 * 128 + j0] + wx1 * sA[i1 * 128 + j1]);
            float bv = wy0 * (wx0 * sB[i0 * 128 + j0] + wx1 * sB[i0 * 128 + j1])
                     + wy1 * (wx0 * sB[i1 * 128 + j0] + wx1 * sB[i1 * 128 + j1]);
            float t = tanhf(0.5f * (bv - av));
            vx[s] = t > 0.f ? t : 0.f;                 // relu(tanh(...))
        }
        xh[i].x = (_Float16)vx[0];
        xh[i].y = (_Float16)vx[1];
    }

    // ---- write x to global (L2-resident for kernel 2) ----
    const size_t base2 = (size_t)bp * 32768 + (size_t)r * 128 + c0;  // h2 units
    h2* gx = (h2*)ws_x;
    {
        alignas(16) h2 obuf[32];
        #pragma unroll
        for (int i = 0; i < 32; ++i) obuf[i] = xh[i];
        #pragma unroll
        for (int v = 0; v < 8; ++v) ((uint4*)(gx + base2))[v] = ((uint4*)obuf)[v];
    }

    // ---- seed cur = -x, borders = 1.0 ----
    const _Float16 ONE = (_Float16)1.0f;
    #pragma unroll
    for (int i = 0; i < 32; ++i) {
        h2 cs = -xh[i];
        if (r == 0 || r == 255) { cs.x = ONE; cs.y = ONE; }
        int cp = c0 + i;
        if (cp == 0)   cs.x = ONE;    // col 0
        if (cp == 127) cs.y = ONE;    // col 255
        buf0[r * 128 + cp] = cs;
    }
    __syncthreads();   // staging reads + seed writes done; loop may clobber buf1

    // ---- 256 iterations: cur = maxpool3x3(cur) * x, all in LDS ----
    const _Float16 NINF = (_Float16)(-__builtin_huge_valf());
    const h2 NEG2 = {NINF, NINF};
    h2* gcur = (h2*)ws_cur;
    for (int t = 0; t < 256; ++t) {
        const h2* src = (t & 1) ? buf1 : buf0;
        h2*       dst = (t & 1) ? buf0 : buf1;
        alignas(16) h2 vm[32];
        bool first = true;
        #pragma unroll
        for (int dr = -1; dr <= 1; ++dr) {
            int rr = r + dr;
            if (rr < 0 || rr > 255) continue;          // -inf pad rows: skip
            const h2* row = src + rr * 128;
            alignas(16) h2 mm[32];
            #pragma unroll
            for (int v = 0; v < 8; ++v)
                ((uint4*)mm)[v] = ((const uint4*)(row + c0))[v];   // ds_load_b128
            h2 L = (c0 > 0)        ? row[c0 - 1]  : NEG2;
            h2 R = (c0 + 32 < 128) ? row[c0 + 32] : NEG2;
            #pragma unroll
            for (int i = 0; i < 32; ++i) {
                h2 le = (i == 0)  ? L : mm[i - 1];
                h2 ri = (i == 31) ? R : mm[i + 1];
                // hm.x = max(le.y, m.x, m.y) ; hm.y = max(m.x, m.y, ri.x)
                h2 aa = {le.y, mm[i].x};
                h2 bb = {mm[i].y, ri.x};
                h2 hm = PKMAX(PKMAX(aa, mm[i]), bb);   // v_pk_max_num_f16
                vm[i] = first ? hm : PKMAX(vm[i], hm);
            }
            first = false;
        }
        #pragma unroll
        for (int i = 0; i < 32; ++i) vm[i] = vm[i] * xh[i];        // v_pk_mul_f16
        h2* drow = dst + r * 128 + c0;
        #pragma unroll
        for (int v = 0; v < 8; ++v) ((uint4*)drow)[v] = ((uint4*)vm)[v];
        if (t == 255) {   // final result lands in this thread's own tile
            #pragma unroll
            for (int v = 0; v < 8; ++v) ((uint4*)(gcur + base2))[v] = ((uint4*)vm)[v];
        }
        __syncthreads();
    }
}

__global__ __launch_bounds__(256)
void k_norm_l1(const _Float16* __restrict__ ws_x, const _Float16* __restrict__ ws_cur,
               float* __restrict__ acc)
{
    const int lane  = threadIdx.x & 31;
    const int wave  = blockIdx.x * (blockDim.x >> 5) + (threadIdx.x >> 5);
    const int px0   = wave << 4;          // 16 pixels per wave
    const int b     = px0 >> 16;          // 65536 pixels per batch
    const int pin   = px0 & 65535;
    const int m     = lane & 15;          // A-matrix row = this lane's pixel
    const int p     = pin + m;
    const int kbase = (lane >> 4) << 3;   // lanes 0-15: K {0..7,16..23}; 16-31: {8..15,24..31}

    // A (16 pixels x 32 channels, fp16) per the CDNA5 16-bit A layout.
    v16h a;
    #pragma unroll
    for (int j = 0; j < 16; ++j) {
        int k = ((j < 8) ? j : j + 8) + kbase;
        a[j] = ws_cur[(((size_t)b * 32 + k) << 16) + p];
    }
    // B's lane layout mirrors A with M<->N, so passing A twice gives D = A*A^T;
    // diag(D)[m] = sum_k cur[pixel m][k]^2.
    v8f c = {0.f, 0.f, 0.f, 0.f, 0.f, 0.f, 0.f, 0.f};
    v8f d = __builtin_amdgcn_wmma_f32_16x16x32_f16(false, a, false, a,
                                                   (short)0, c, false, false);
    // diag(m) lives at (VGPR m&7, lane m) for m<8 and (VGPR m-8, lane m+16)
    // for m>=8; equivalently the owner lane s in {0..7, 24..31} holds d[s&7].
    // So every lane exposes d[lane&7] and one bpermute fetches the diagonal.
    float vv = d[0];
    #pragma unroll
    for (int j = 1; j < 8; ++j) vv = ((lane & 7) == j) ? d[j] : vv;
    const int srcl = (m < 8) ? m : m + 16;
    float s  = __shfl(vv, srcl, 32);
    float rn = 1.0f / fmaxf(sqrtf(s), 1e-12f);

    // |x - (cur/maxnorm)*x| = x * |1 - cur*rn|   (x >= 0 from relu)
    float lsum = 0.f;
    #pragma unroll
    for (int j = 0; j < 16; ++j) {
        int k = ((j < 8) ? j : j + 8) + kbase;
        float cf = (float)a[j];
        float xf = (float)ws_x[(((size_t)b * 32 + k) << 16) + p];
        lsum += xf * fabsf(1.0f - cf * rn);
    }
    #pragma unroll
    for (int off = 16; off; off >>= 1) lsum += __shfl_xor(lsum, off, 32);
    if (lane == 0) atomicAdd(acc, lsum);
}

__global__ void k_init(float* acc) { *acc = 0.f; }

__global__ void k_finalize(const float* __restrict__ acc, _Float16* __restrict__ out)
{
    float v = *acc;
    float r = fminf(v, 65504.f);
    out[0] = (_Float16)(__builtin_isnan(v) ? 0.f : r);
}

extern "C" void kernel_launch(void* const* d_in, const int* in_sizes, int n_in,
                              void* d_out, int out_size, void* d_ws, size_t ws_size,
                              hipStream_t stream)
{
    (void)in_sizes; (void)n_in; (void)out_size; (void)ws_size;
    const float* input = (const float*)d_in[0];

    const size_t NELEM = (size_t)16 * 32 * 256 * 256;     // 33,554,432
    _Float16* ws_x   = (_Float16*)d_ws;
    _Float16* ws_cur = ws_x + NELEM;                      // +64 MiB
    float*    acc    = (float*)((char*)d_ws + 2 * NELEM * sizeof(_Float16)); // +128 MiB

    k_init<<<1, 1, 0, stream>>>(acc);
    // 512 workgroups (one per (batch,pair) image), 256KB dynamic LDS each.
    k_seed_iterate<<<512, 1024, 2 * 256 * 128 * sizeof(h2), stream>>>(input, ws_x, ws_cur);
    // 16*65536 pixels / (16 px/wave * 8 waves/block) = 8192 blocks.
    k_norm_l1<<<8192, 256, 0, stream>>>(ws_x, ws_cur, acc);
    k_finalize<<<1, 1, 0, stream>>>(acc, (_Float16*)d_out);
}
